// BiLSTMCRF_52450140619117
// MI455X (gfx1250) — matrix operand order repaired and checked by reading, not compiled, not run on hardware
//
#include <hip/hip_runtime.h>
#include <hip/hip_bf16.h>

// ---------------------------------------------------------------------------
// BiLSTM-CRF forward loss for MI455X (gfx1250, wave32, WMMA).
//
// Pipeline (all on `stream`):
//   1. embed_char_kernel : word emb + 1-step char LSTM -> x0 in A-fragment bf16 layout
//   2. gemm_in_kernel<6> : xg0 = x0 @ W0ih^T + b0  (bf16 WMMA, f32 acc, C-fragment layout)
//   3. lstm_scan_kernel  : layer-0 fwd/bwd scans (WMMA recurrence, Whh in LDS) -> x1 frags
//   4. gemm_in_kernel<8> : xg1 = h0cat @ W1ih^T + b1
//   5. lstm_scan_kernel  : layer-1 scans -> h1 plain f32
//   6. emissions_kernel  : em = h1 @ Wtag^T + btag
//   7. crf_kernel        : CRF NLL (sum) -> d_out[0]
// ---------------------------------------------------------------------------

typedef __attribute__((ext_vector_type(16))) __bf16 v16bf;
typedef __attribute__((ext_vector_type(8)))  float  v8f;

#define BATCH 32

__device__ __forceinline__ float sigf(float x) { return 1.f / (1.f + __expf(-x)); }

__device__ __forceinline__ float tanh_fast(float x) {
  x = fminf(fmaxf(x, -15.f), 15.f);
  float e = __expf(-2.f * x);
  return (1.f - e) / (1.f + e);
}

__device__ __forceinline__ float lse3(float a, float b, float c) {
  float m = fmaxf(a, fmaxf(b, c));
  return m + __logf(__expf(a - m) + __expf(b - m) + __expf(c - m));
}

// Index into an A-matrix (16x32 bf16) fragment buffer laid out so each lane's
// 16 elements are contiguous:  idx = ((mtile*KT + ktile)*32 + lane)*16 + e.
// Contiguous K order (matches B fragments): lanes 0-15 (M=0-15) hold K 0..15;
// lanes 16-31 hold K 16..31.
__device__ __forceinline__ int afrag_idx(int KT, int b, int feat) {
  int mt = b >> 4, row = b & 15;
  int kt = feat >> 5, kk = feat & 31;
  int half, e;
  half = kk >> 4; e = kk & 15;
  return ((mt * KT + kt) * 32 + ((half << 4) | row)) * 16 + e;
}

// ---------------------------------------------------------------------------
// 1. Embedding + 1-step char BiLSTM.  One thread per (t, b) token.
// ---------------------------------------------------------------------------
__global__ void embed_char_kernel(const int* __restrict__ seq,
                                  const float* __restrict__ word_emb,
                                  const float* __restrict__ char_emb,
                                  const float* __restrict__ cWih,
                                  const float* __restrict__ cb,
                                  __bf16* __restrict__ x0, int S) {
  __shared__ float sW[2 * 128 * 32];
  for (int i = threadIdx.x; i < 2 * 128 * 32; i += blockDim.x) sW[i] = cWih[i];
  __syncthreads();

  int gid = blockIdx.x * blockDim.x + threadIdx.x;
  if (gid >= S * BATCH) return;
  int t = gid >> 5, b = gid & 31;
  int s = seq[(size_t)b * S + t] & 255;

  __bf16* xt = x0 + (size_t)t * 6144;          // 2*6 frags * 32 lanes * 16
  const float* we = word_emb + (size_t)s * 128;
  for (int f = 0; f < 128; ++f) xt[afrag_idx(6, b, f)] = (__bf16)we[f];

  float ce[32];
  const float* cep = char_emb + (size_t)s * 32;
  for (int j = 0; j < 32; ++j) ce[j] = cep[j];

  for (int d = 0; d < 2; ++d) {
    const float* W  = sW + d * 128 * 32;
    const float* bb = cb + d * 128;
    for (int u = 0; u < 32; ++u) {
      float gi = bb[u], gg = bb[64 + u], go = bb[96 + u];   // f-gate * c0 == 0
      for (int j = 0; j < 32; ++j) {
        float c = ce[j];
        gi += W[u * 32 + j] * c;
        gg += W[(64 + u) * 32 + j] * c;
        go += W[(96 + u) * 32 + j] * c;
      }
      float cc = sigf(gi) * tanh_fast(gg);
      float h  = sigf(go) * tanh_fast(cc);
      xt[afrag_idx(6, b, 128 + d * 32 + u)] = (__bf16)h;
    }
  }
}

// ---------------------------------------------------------------------------
// 2/4. Input projection GEMM: xg[dir][t][b][512] = x[t][b][:] @ Wih^T + b.
//   Block = one timestep (blockIdx.x), dir = blockIdx.y, 8 wave32 waves.
//   Wave w computes gate N-tiles {w, w+8, w+16, w+24} for both batch M-tiles.
//   Output stored in C/D-fragment layout for direct use as the scan's WMMA C.
// ---------------------------------------------------------------------------
template <int KT>
__global__ __launch_bounds__(256, 1)
void gemm_in_kernel(const __bf16* __restrict__ xfrag,
                    const float* __restrict__ W,
                    const float* __restrict__ bias,
                    float* __restrict__ xg, int S) {
  const int Kin = KT * 32;
  int t = blockIdx.x, dir = blockIdx.y;
  const float* Wd = W + (size_t)dir * 512 * Kin;
  const float* bd = bias + dir * 512;
  float* xgd = xg + (size_t)dir * (size_t)S * 16384;

  int wv = threadIdx.x >> 5, lane = threadIdx.x & 31;
  int half = lane >> 4, nl = lane & 15;

  const __bf16* xb = xfrag + (size_t)t * (size_t)(2 * KT * 512) + lane * 16;
  float* outb = xgd + (size_t)t * 16384 + lane * 8;

#pragma unroll
  for (int m = 0; m < 2; ++m) {
    v16bf A[KT];
#pragma unroll
    for (int k = 0; k < KT; ++k)
      A[k] = *(const v16bf*)(xb + (m * KT + k) * 512);

#pragma unroll
    for (int gt = 0; gt < 4; ++gt) {
      int ntile = wv + 8 * gt;
      int gate  = ntile * 16 + nl;
      float bv  = bd[gate];
      v8f acc;
#pragma unroll
      for (int r = 0; r < 8; ++r) acc[r] = bv;
#pragma unroll
      for (int k = 0; k < KT; ++k) {
        // B fragment (32x16 bf16): lane n holds K 0..15, lane n+16 holds K 16..31.
        v16bf Bf;
        const float* wp = Wd + (size_t)gate * Kin + k * 32 + half * 16;
#pragma unroll
        for (int e = 0; e < 16; ++e) Bf[e] = (__bf16)wp[e];
        acc = __builtin_amdgcn_wmma_f32_16x16x32_bf16(false, A[k], false, Bf,
                                                      (short)0, acc, false, false);
      }
      *(v8f*)(outb + (m * 32 + ntile) * 256) = acc;
    }
  }
}

// ---------------------------------------------------------------------------
// 3/5. Sequential LSTM scan (one block per direction, 8 waves).
//   Per step: G = xg[t] (C frags, global) + h @ Whh^T (WMMA, Whh in LDS).
//   mode 0: write h as bf16 A-fragments (KT=8, feat = dir*128+u) for next GEMM.
//   mode 1: write h plain f32 [t][b][256] for emissions.
//   Dynamic LDS: 128KB Whh B-fragments + 8KB h A-fragments.
// ---------------------------------------------------------------------------
__global__ __launch_bounds__(256, 1)
void lstm_scan_kernel(const float* __restrict__ xg,
                      const float* __restrict__ Whh,
                      void* __restrict__ hout, int S, int mode) {
  int dir = blockIdx.x;
  extern __shared__ char smem[];
  __bf16* whhL = (__bf16*)smem;                // 65536 elems = 128KB
  __bf16* hL   = (__bf16*)(smem + 131072);     // 4096 elems  = 8KB

  const float* Wh  = Whh + (size_t)dir * 512 * 128;
  const float* xgd = xg + (size_t)dir * (size_t)S * 16384;

  int tid = threadIdx.x;
  int wv = tid >> 5, lane = tid & 31, half = lane >> 4, nl = lane & 15;

  // Pre-swizzle Whh^T into B-fragment layout in LDS (gathers are contiguous in K).
  for (int rr = 0; rr < 16; ++rr) {
    int rid = tid + rr * 256;                  // 4096 fragment-rows total
    int frag = rid >> 5, ln = rid & 31;
    int ktile = frag >> 5, ntile = frag & 31;
    int gate = ntile * 16 + (ln & 15);
    const float* wp = Wh + (size_t)gate * 128 + ktile * 32 + (ln >> 4) * 16;
    __bf16* dst = whhL + (frag * 32 + ln) * 16;
    for (int e = 0; e < 16; ++e) dst[e] = (__bf16)wp[e];
  }
  for (int i = tid; i < 4096; i += 256) hL[i] = (__bf16)0.f;  // h0 = 0

  // Lane-invariant base pointers -> inner accesses use immediate offsets.
  const __bf16* hA   = hL + lane * 16;            // + (m*4+k)*512
  const __bf16* whhB = whhL + lane * 16;          // + (k*32+ntile)*512
  int u = wv * 16 + nl;                           // hidden unit this lane owns
  int hl_idx0 = afrag_idx(4, half * 8, u);        // batch = m*16 + half*8 + r
  int x1_idx0 = 0, h1_idx0 = 0;
  if (mode == 0) x1_idx0 = afrag_idx(8, half * 8, 128 * dir + u);
  else           h1_idx0 = (half * 8) * 256 + 128 * dir + u;

  float cst[2][8];
#pragma unroll
  for (int m = 0; m < 2; ++m)
#pragma unroll
    for (int r = 0; r < 8; ++r) cst[m][r] = 0.f;              // c0 = 0
  __syncthreads();

  for (int step = 0; step < S; ++step) {
    int t = dir ? (S - 1 - step) : step;
    const float* xgt = xgd + (size_t)t * 16384 + lane * 8;

    if (step + 1 < S) {
      const float* xgn = dir ? (xgt - 16384) : (xgt + 16384);
      __builtin_prefetch(xgn + wv * 256, 0, 1);
      __builtin_prefetch(xgn + (32 + wv) * 256, 0, 1);
    }

    float hv[2][8];
#pragma unroll
    for (int m = 0; m < 2; ++m) {
      v16bf A[4];
#pragma unroll
      for (int k = 0; k < 4; ++k)
        A[k] = *(const v16bf*)(hA + (m * 4 + k) * 512);

      v8f acc[4];
#pragma unroll
      for (int gt = 0; gt < 4; ++gt) {
        int ntile = wv + 8 * gt;
        acc[gt] = *(const v8f*)(xgt + (m * 32 + ntile) * 256);
#pragma unroll
        for (int k = 0; k < 4; ++k) {
          v16bf Bf = *(const v16bf*)(whhB + (k * 32 + ntile) * 512);
          acc[gt] = __builtin_amdgcn_wmma_f32_16x16x32_bf16(false, A[k], false, Bf,
                                                            (short)0, acc[gt], false, false);
        }
      }
#pragma unroll
      for (int r = 0; r < 8; ++r) {                 // PyTorch gate order i,f,g,o
        float ig = sigf(acc[0][r]);
        float fg = sigf(acc[1][r]);
        float gg = tanh_fast(acc[2][r]);
        float og = sigf(acc[3][r]);
        float c  = fg * cst[m][r] + ig * gg;
        cst[m][r] = c;
        hv[m][r]  = og * tanh_fast(c);
      }
    }
    __syncthreads();                                // all A-reads of old h done

#pragma unroll
    for (int m = 0; m < 2; ++m)
#pragma unroll
      for (int r = 0; r < 8; ++r) {
        float v = hv[m][r];
        // afrag layouts: +m*16 on batch adds mt*KT*512; +r on batch adds r within rows.
        hL[hl_idx0 + m * 4 * 512 + r * 16] = (__bf16)v;
        if (mode == 0)
          ((__bf16*)hout)[(size_t)t * 8192 + x1_idx0 + m * 8 * 512 + r * 16] = (__bf16)v;
        else
          ((float*)hout)[(size_t)t * 8192 + h1_idx0 + m * 16 * 256 + r * 256] = v;
      }
    __syncthreads();                                // new h visible next step
  }
}

// ---------------------------------------------------------------------------
// 6. Emissions: em[t,b,tag] = h1[t,b,:] . Wtag[tag,:] + btag.  (N=3: VALU dots)
// ---------------------------------------------------------------------------
__global__ void emissions_kernel(const float* __restrict__ h1,
                                 const float* __restrict__ Wtag,
                                 const float* __restrict__ btag,
                                 float* __restrict__ em, int total) {
  int gid = blockIdx.x * blockDim.x + threadIdx.x;
  if (gid >= total) return;
  const float* h = h1 + (size_t)gid * 256;
  float s0 = btag[0], s1 = btag[1], s2 = btag[2];
  for (int j = 0; j < 256; ++j) {
    float v = h[j];
    s0 += v * Wtag[j];
    s1 += v * Wtag[256 + j];
    s2 += v * Wtag[512 + j];
  }
  float* e = em + (size_t)gid * 3;
  e[0] = s0; e[1] = s1; e[2] = s2;
}

// ---------------------------------------------------------------------------
// 7. CRF NLL (sum reduction). One wave, one lane per batch row.
// ---------------------------------------------------------------------------
__global__ void crf_kernel(const int* __restrict__ tags,
                           const int* __restrict__ masks,
                           const float* __restrict__ em,
                           const float* __restrict__ st,
                           const float* __restrict__ en,
                           const float* __restrict__ trans,
                           float* __restrict__ out, int S) {
  __shared__ float red[BATCH];
  int b = threadIdx.x;
  float tr[9];
  for (int i = 0; i < 9; ++i) tr[i] = trans[i];

  int   tprev = tags[(size_t)b * S];
  float a0 = st[0] + em[(size_t)b * 3 + 0];
  float a1 = st[1] + em[(size_t)b * 3 + 1];
  float a2 = st[2] + em[(size_t)b * 3 + 2];
  float num = st[tprev] + em[(size_t)b * 3 + tprev];
  int msum = (masks[(size_t)b * S] != 0) ? 1 : 0;

  for (int t = 1; t < S; ++t) {
    int mi = masks[(size_t)b * S + t];
    float m = (float)mi;
    int tt = tags[(size_t)b * S + t];
    const float* e = em + ((size_t)t * BATCH + b) * 3;
    num += (tr[tprev * 3 + tt] + e[tt]) * m;
    tprev = tt;
    float n0 = lse3(a0 + tr[0], a1 + tr[3], a2 + tr[6]) + e[0];
    float n1 = lse3(a0 + tr[1], a1 + tr[4], a2 + tr[7]) + e[1];
    float n2 = lse3(a0 + tr[2], a1 + tr[5], a2 + tr[8]) + e[2];
    if (mi != 0) { a0 = n0; a1 = n1; a2 = n2; msum += 1; }
  }
  int lt = tags[(size_t)b * S + (msum - 1)];
  num += en[lt];
  float den = lse3(a0 + en[0], a1 + en[1], a2 + en[2]);
  red[b] = den - num;
  __syncthreads();
  if (b == 0) {
    float s = 0.f;
    for (int i = 0; i < BATCH; ++i) s += red[i];
    out[0] = s;
  }
}

// ---------------------------------------------------------------------------
extern "C" void kernel_launch(void* const* d_in, const int* in_sizes, int n_in,
                              void* d_out, int out_size, void* d_ws, size_t ws_size,
                              hipStream_t stream) {
  (void)n_in; (void)out_size; (void)ws_size;
  const int S = in_sizes[0] / BATCH;   // 2048

  const int*   seq      = (const int*)d_in[0];
  const int*   tags     = (const int*)d_in[1];
  const int*   masks    = (const int*)d_in[2];
  const float* word_emb = (const float*)d_in[3];
  const float* char_emb = (const float*)d_in[4];
  const float* cWih     = (const float*)d_in[5];
  const float* cb       = (const float*)d_in[6];
  const float* l0Wih    = (const float*)d_in[7];
  const float* l0Whh    = (const float*)d_in[8];
  const float* l0b      = (const float*)d_in[9];
  const float* l1Wih    = (const float*)d_in[10];
  const float* l1Whh    = (const float*)d_in[11];
  const float* l1b      = (const float*)d_in[12];
  const float* Wtag     = (const float*)d_in[13];
  const float* btag     = (const float*)d_in[14];
  const float* st       = (const float*)d_in[15];
  const float* en       = (const float*)d_in[16];
  const float* trans    = (const float*)d_in[17];

  char* ws = (char*)d_ws;
  size_t off = 0;
  __bf16* x0 = (__bf16*)(ws + off); off += (size_t)S * 6144 * sizeof(__bf16);     // layer0 A-frags
  __bf16* x1 = (__bf16*)(ws + off); off += (size_t)S * 8192 * sizeof(__bf16);     // layer1 A-frags
  float*  xg = (float*)(ws + off);  off += (size_t)2 * S * 16384 * sizeof(float); // gate C-frags (reused per layer)
  float*  h1 = (float*)(ws + off);  off += (size_t)S * BATCH * 256 * sizeof(float);
  float*  em = (float*)(ws + off);  off += (size_t)S * BATCH * 3 * sizeof(float);

  const int tokBlocks = (S * BATCH + 255) / 256;
  const size_t scanLds = 131072 + 8192;

  embed_char_kernel<<<tokBlocks, 256, 0, stream>>>(seq, word_emb, char_emb, cWih, cb, x0, S);
  gemm_in_kernel<6><<<dim3(S, 2), 256, 0, stream>>>(x0, l0Wih, l0b, xg, S);
  lstm_scan_kernel<<<2, 256, scanLds, stream>>>(xg, l0Whh, (void*)x1, S, 0);
  gemm_in_kernel<8><<<dim3(S, 2), 256, 0, stream>>>(x1, l1Wih, l1b, xg, S);
  lstm_scan_kernel<<<2, 256, scanLds, stream>>>(xg, l1Whh, (void*)h1, S, 1);
  emissions_kernel<<<tokBlocks, 256, 0, stream>>>(h1, Wtag, btag, em, S * BATCH);
  crf_kernel<<<1, BATCH, 0, stream>>>(tags, masks, em, st, en, trans, (float*)d_out, S);
}
